// GATop_42872363548900
// MI455X (gfx1250) — compile-verified, matrix operand order checked by reading, not run on hardware
//
#include <hip/hip_runtime.h>

// ---------------------------------------------------------------------------
// GAT-op forward for MI455X (gfx1250, wave32, WMMA + Tensor Data Mover).
// All GEMM-shaped math -> v_wmma_f32_16x16x32_f16.
// Weights are fragment-swizzled to f16 once (workspace), then DMA'd into LDS
// per workgroup with TENSOR_LOAD_TO_LDS (TENSORcnt / s_wait_tensorcnt), so the
// hot WMMA loops read A- and B-fragments from LDS (ds_load_b128) only.
// ---------------------------------------------------------------------------

typedef __attribute__((ext_vector_type(16))) _Float16 v16h;
typedef __attribute__((ext_vector_type(8)))  _Float16 v8h;
typedef __attribute__((ext_vector_type(8)))  float    v8f;
typedef __attribute__((ext_vector_type(4)))  unsigned u32x4;
typedef __attribute__((ext_vector_type(8)))  unsigned u32x8;

#define B_     16
#define NOP_   512
#define NMA_   64
#define HID_   256
#define OUT_   128

__device__ inline void lds_fence() {
#if defined(__has_builtin)
#if __has_builtin(__builtin_amdgcn_s_wait_dscnt)
  __builtin_amdgcn_s_wait_dscnt(0);
#else
  __builtin_amdgcn_fence(__ATOMIC_ACQ_REL, "workgroup");
#endif
#else
  __builtin_amdgcn_fence(__ATOMIC_ACQ_REL, "workgroup");
#endif
}

__device__ inline v8f zero8() {
  v8f z;
#pragma unroll
  for (int i = 0; i < 8; ++i) z[i] = 0.0f;
  return z;
}

__device__ inline unsigned lds_off(const void* p) {
  // LDS aperture: addr[31:0] is the byte offset within the wave's LDS.
  return (unsigned)(uintptr_t)p;
}

// ---------------------------------------------------------------------------
// Tensor Data Mover: contiguous global -> LDS copy as 1-row 2D tiles.
// D# per CDNA5 ISA: group0 = {count=1, lds_addr,
// global_addr[56:0], type=2}; group1 = {wg_mask=0, data_size=4B, dims}.
// Issued by one wave (EXEC ignored); tracked by TENSORcnt.
// ---------------------------------------------------------------------------
__device__ inline void tdm_copy_g2l(unsigned lds_bytes, const void* g,
                                    unsigned nbytes) {
  unsigned ndw_total = nbytes >> 2;
  unsigned done = 0;
  while (done < ndw_total) {
    unsigned ndw = ndw_total - done;
    if (ndw > 32768u) ndw = 32768u;          // tile_dim0 is 16-bit
    unsigned long long ga = (unsigned long long)g + ((unsigned long long)done << 2);
    u32x4 g0;
    g0.x = 1u;                                // count=1 (valid user D#)
    g0.y = lds_bytes + (done << 2);           // lds_addr (bytes)
    g0.z = (unsigned)ga;                      // global_addr[31:0]
    g0.w = (unsigned)((ga >> 32) & 0x01FFFFFFull) | 0x80000000u; // [56:32]|type=2
    u32x8 g1;
    g1[0] = 2u << 16;                         // data_size = 4 bytes
    g1[1] = (ndw & 0xFFFFu) << 16;            // tensor_dim0[15:0]
    g1[2] = (ndw >> 16) | (1u << 16);         // tensor_dim0[31:16] | tensor_dim1=1
    g1[3] = (ndw & 0xFFFFu) << 16;            // tile_dim0 = ndw
    g1[4] = 1u;                               // tile_dim1 = 1, tile_dim2 = 0
    g1[5] = ndw;                              // tensor_dim0_stride (don't-care, 1 row)
    g1[6] = 0u;
    g1[7] = 0u;
    asm volatile("tensor_load_to_lds %0, %1" :: "s"(g0), "s"(g1) : "memory");
    done += ndw;
  }
}

// A-fragment (16x32 f16, MxK) from a row-major f16 tile in LDS.
__device__ inline v16h load_a_frag(const _Float16* tile, int strideH) {
  const int lane = threadIdx.x & 31;
  const int row  = lane & 15;
  const int koff = (lane & 16) ? 8 : 0;
  const _Float16* p = tile + row * strideH + koff;
  v8h lo = *(const v8h*)(p);        // K = koff .. koff+7   (ds_load_b128)
  v8h hi = *(const v8h*)(p + 16);   // K = koff+16 .. koff+23
  v16h a;
#pragma unroll
  for (int i = 0; i < 8; ++i) { a[i] = lo[i]; a[8 + i] = hi[i]; }
  return a;
}

// B-fragment from a pre-swizzled tile (512 halves, lane-major).
__device__ inline v16h load_b_frag(const _Float16* tile) {
  const int lane = threadIdx.x & 31;
  return *(const v16h*)(tile + lane * 16);   // 32B/lane -> ds_load_b128 x2
}

// ---------------------------------------------------------------------------
// Swizzle K x N f32 (K zero-padded to KPAD) into B-operand fragment tiles.
// ---------------------------------------------------------------------------
__global__ __launch_bounds__(256) void swizzle_weight(
    const float* __restrict__ W, _Float16* __restrict__ dst,
    int K, int N, int KPAD) {
  int total = KPAD * N;
  for (int i = blockIdx.x * blockDim.x + threadIdx.x; i < total;
       i += gridDim.x * blockDim.x) {
    int k = i / N, n = i % N;
    float v = (k < K) ? W[(size_t)k * N + n] : 0.0f;
    int kt = k >> 5, kk = k & 31, nt = n >> 4, nn = n & 15;
    int lane = nn + (((kk >> 3) & 1) << 4);
    int e    = (kk & 7) + ((kk & 16) ? 8 : 0);
    dst[((size_t)(kt * (N >> 4) + nt) << 9) + lane * 16 + e] = (_Float16)v;
  }
}

// ---------------------------------------------------------------------------
// pre_x / sub_x = adj @ op  (per batch: 512x512 {0,1} @ 512x16).
// op-tiles DMA'd to LDS; two interleaved accumulators avoid the WMMA->WMMA
// RAW hazard NOPs seen with a single chain.
// ---------------------------------------------------------------------------
__global__ __launch_bounds__(128) void gemm_adj_kernel(
    const int* __restrict__ adj, const _Float16* __restrict__ optiles,
    float* __restrict__ Xout) {
  __shared__ _Float16 opt_s[16 * 512];
  const int wave = blockIdx.x * 4 + (threadIdx.x >> 5);
  const int b  = wave >> 5;       // uniform per block (4 waves, 32 tiles/batch)
  const int mt = wave & 31;
  const int lane = threadIdx.x & 31;
  const int col = lane & 15;
  const int mo  = (lane & 16) ? 8 : 0;
  const int k0  = (lane & 16) ? 8 : 0;

  if (threadIdx.x < 32) {
    tdm_copy_g2l(lds_off(opt_s), optiles + (((size_t)b * 16) << 9),
                 16 * 512 * 2);
    __builtin_amdgcn_s_wait_tensorcnt(0);
  }
  __syncthreads();

  const int* arow = adj + ((size_t)b * NOP_ + mt * 16 + (lane & 15)) * NOP_;
  v8f c0 = zero8(), c1 = zero8();
  for (int kt = 0; kt < 16; kt += 2) {
#pragma unroll
    for (int u = 0; u < 2; ++u) {
      v16h a;
#pragma unroll
      for (int e = 0; e < 16; ++e) {
        int K = (kt + u) * 32 + k0 + (e & 7) + ((e & 8) ? 16 : 0);
        a[e] = (_Float16)(float)arow[K];
      }
      v16h bf = load_b_frag(opt_s + ((kt + u) << 9));
      if (u == 0)
        c0 = __builtin_amdgcn_wmma_f32_16x16x32_f16(false, a, false, bf,
                                                    (short)0, c0, false, false);
      else
        c1 = __builtin_amdgcn_wmma_f32_16x16x32_f16(false, a, false, bf,
                                                    (short)0, c1, false, false);
    }
  }
  float* yrow = Xout + ((size_t)b * NOP_ + mt * 16) * 16;
#pragma unroll
  for (int r = 0; r < 8; ++r) yrow[(r + mo) * 16 + col] = c0[r] + c1[r];
}

// ---------------------------------------------------------------------------
// 3-layer MLP on a 16-row tile; weights already resident in LDS.
// ---------------------------------------------------------------------------
template <int KT1>
__device__ inline void mlp_forward(_Float16* hbuf,
    const _Float16* w1f, const _Float16* w2f, const _Float16* w3f,
    const float* __restrict__ b1, const float* __restrict__ b2,
    const float* __restrict__ b3, v8f out[8]) {
  const int lane = threadIdx.x & 31;
  const int col  = lane & 15;
  const int mo   = (lane & 16) ? 8 : 0;

  // ---- layer 1: (16 x 32*KT1) @ (32*KT1 x 256), ELU ----
  {
    v8f acc[16];
#pragma unroll
    for (int nt = 0; nt < 16; ++nt) acc[nt] = zero8();
    for (int kt = 0; kt < KT1; ++kt) {
      v16h a = load_a_frag(hbuf + kt * 32, KT1 * 32);
#pragma unroll
      for (int nt = 0; nt < 16; ++nt) {
        v16h bf = load_b_frag(w1f + ((size_t)((kt << 4) + nt) << 9));
        acc[nt] = __builtin_amdgcn_wmma_f32_16x16x32_f16(false, a, false, bf,
                                       (short)0, acc[nt], false, false);
      }
    }
#pragma unroll
    for (int nt = 0; nt < 16; ++nt) {
      float bias = b1[(nt << 4) + col];
#pragma unroll
      for (int r = 0; r < 8; ++r) {
        float x = acc[nt][r] + bias;
        x = (x > 0.0f) ? x : expm1f(x);                 // ELU
        hbuf[(r + mo) * 256 + (nt << 4) + col] = (_Float16)x;
      }
    }
  }
  lds_fence();

  // ---- layer 2: (16 x 256) @ (256 x 256), ELU ----
  {
    v8f acc[16];
#pragma unroll
    for (int nt = 0; nt < 16; ++nt) acc[nt] = zero8();
    for (int kt = 0; kt < 8; ++kt) {
      v16h a = load_a_frag(hbuf + kt * 32, 256);
#pragma unroll
      for (int nt = 0; nt < 16; ++nt) {
        v16h bf = load_b_frag(w2f + ((size_t)((kt << 4) + nt) << 9));
        acc[nt] = __builtin_amdgcn_wmma_f32_16x16x32_f16(false, a, false, bf,
                                       (short)0, acc[nt], false, false);
      }
    }
    lds_fence();
#pragma unroll
    for (int nt = 0; nt < 16; ++nt) {
      float bias = b2[(nt << 4) + col];
#pragma unroll
      for (int r = 0; r < 8; ++r) {
        float x = acc[nt][r] + bias;
        x = (x > 0.0f) ? x : expm1f(x);
        hbuf[(r + mo) * 256 + (nt << 4) + col] = (_Float16)x;
      }
    }
  }
  lds_fence();

  // ---- layer 3: (16 x 256) @ (256 x 128), +bias ----
  {
#pragma unroll
    for (int nt = 0; nt < 8; ++nt) out[nt] = zero8();
    for (int kt = 0; kt < 8; ++kt) {
      v16h a = load_a_frag(hbuf + kt * 32, 256);
#pragma unroll
      for (int nt = 0; nt < 8; ++nt) {
        v16h bf = load_b_frag(w3f + ((size_t)((kt << 3) + nt) << 9));
        out[nt] = __builtin_amdgcn_wmma_f32_16x16x32_f16(false, a, false, bf,
                                       (short)0, out[nt], false, false);
      }
    }
#pragma unroll
    for (int nt = 0; nt < 8; ++nt) {
      float bias = b3[(nt << 4) + col];
#pragma unroll
      for (int r = 0; r < 8; ++r) out[nt][r] += bias;
    }
  }
}

// ---------------------------------------------------------------------------
// mlp0 + masked-mean over NMA. 8 waves/block, one wave per op node.
// Weights (208 KB f16 fragments) DMA'd to LDS once per block via TDM.
// ---------------------------------------------------------------------------
__global__ __launch_bounds__(256) void mlp0_agg_kernel(
    const int* __restrict__ ma_adj, const float* __restrict__ ma,
    const float* __restrict__ eg,
    const _Float16* __restrict__ w1f, const _Float16* __restrict__ w2f,
    const _Float16* __restrict__ w3f,
    const float* __restrict__ b1, const float* __restrict__ b2,
    const float* __restrict__ b3, float* __restrict__ ma_in) {
  __shared__ _Float16 w1_s[16 * 512];          // 16 KB
  __shared__ _Float16 w2_s[128 * 512];         // 128 KB
  __shared__ _Float16 w3_s[64 * 512];          // 64 KB
  __shared__ _Float16 hbuf_all[8][16 * 256];   // 64 KB
  const int w = threadIdx.x >> 5, lane = threadIdx.x & 31;
  _Float16* hbuf = hbuf_all[w];
  const int opg = blockIdx.x * 8 + w;
  const int b = opg >> 9, i = opg & (NOP_ - 1);

  if (threadIdx.x < 32) {
    tdm_copy_g2l(lds_off(w1_s), w1f, 16 * 512 * 2);
    tdm_copy_g2l(lds_off(w2_s), w2f, 128 * 512 * 2);
    tdm_copy_g2l(lds_off(w3_s), w3f, 64 * 512 * 2);
    __builtin_amdgcn_s_wait_tensorcnt(0);
  }
  __syncthreads();

  float psum[8], pcnt[8];
#pragma unroll
  for (int nt = 0; nt < 8; ++nt) { psum[nt] = 0.0f; pcnt[nt] = 0.0f; }

  for (int mt = 0; mt < 4; ++mt) {
    if (lane < 16) {
      int j = mt * 16 + lane;
      float adjv = (float)ma_adj[((size_t)b * NOP_ + i) * NMA_ + j];
      const float* mar = ma + ((size_t)b * NMA_ + j) * 8;
      const float* egr = eg + (((size_t)b * NOP_ + i) * NMA_ + j) * 8;
      __builtin_prefetch(egr + 16 * 8, 0, 1);   // next m-tile's eg rows
#pragma unroll
      for (int t = 0; t < 8; ++t)  hbuf[lane * 32 + t]      = (_Float16)(adjv * mar[t]);
#pragma unroll
      for (int t = 0; t < 8; ++t)  hbuf[lane * 32 + 8 + t]  = (_Float16)(adjv * egr[t]);
#pragma unroll
      for (int t = 0; t < 16; ++t) hbuf[lane * 32 + 16 + t] = (_Float16)0.0f;
    }
    lds_fence();

    v8f out[8];
    mlp_forward<1>(hbuf, w1_s, w2_s, w3_s, b1, b2, b3, out);

#pragma unroll
    for (int nt = 0; nt < 8; ++nt) {
#pragma unroll
      for (int r = 0; r < 8; ++r) {
        float v = out[nt][r];
        psum[nt] += v;
        pcnt[nt] += (v != 0.0f) ? 1.0f : 0.0f;
      }
    }
    lds_fence();
  }

  // Combine the two row-halves (lane l <-> l^16 hold the same column).
#pragma unroll
  for (int nt = 0; nt < 8; ++nt) {
    psum[nt] += __shfl_xor(psum[nt], 16, 32);
    pcnt[nt] += __shfl_xor(pcnt[nt], 16, 32);
  }
  if (lane < 16) {
    float* dst = ma_in + ((size_t)b * NOP_ + i) * OUT_;
#pragma unroll
    for (int nt = 0; nt < 8; ++nt) {
      float c = pcnt[nt];
      dst[(nt << 4) + lane] = (c > 0.0f) ? (psum[nt] / c) : 0.0f;
    }
  }
}

// ---------------------------------------------------------------------------
// Generic row-MLP: mlp1/2/3 (KT1=1, 8 waves) and mlp4 (KT1=4, 4 waves).
// ---------------------------------------------------------------------------
template <int KT1, int NW>
__global__ __launch_bounds__(NW * 32) void mlp_rows_kernel(
    const float* __restrict__ X, const _Float16* __restrict__ w1f,
    const _Float16* __restrict__ w2f, const _Float16* __restrict__ w3f,
    const float* __restrict__ b1, const float* __restrict__ b2,
    const float* __restrict__ b3, float* __restrict__ Y, int din) {
  __shared__ _Float16 w1_s[KT1 * 16 * 512];
  __shared__ _Float16 w2_s[128 * 512];
  __shared__ _Float16 w3_s[64 * 512];
  __shared__ _Float16 hbuf_all[NW][16 * 256];
  const int w = threadIdx.x >> 5, lane = threadIdx.x & 31;
  _Float16* hbuf = hbuf_all[w];
  const int tile = blockIdx.x * NW + w;
  const int KPAD = KT1 * 32;

  if (threadIdx.x < 32) {
    tdm_copy_g2l(lds_off(w1_s), w1f, KT1 * 16 * 512 * 2);
    tdm_copy_g2l(lds_off(w2_s), w2f, 128 * 512 * 2);
    tdm_copy_g2l(lds_off(w3_s), w3f, 64 * 512 * 2);
    __builtin_amdgcn_s_wait_tensorcnt(0);
  }
  __syncthreads();

  if (lane < 16) {
    const float* xr = X + ((size_t)tile * 16 + lane) * din;
    __builtin_prefetch(xr + din, 0, 1);
    for (int k = 0; k < KPAD; ++k)
      hbuf[lane * KPAD + k] = (k < din) ? (_Float16)xr[k] : (_Float16)0.0f;
  }
  lds_fence();

  v8f out[8];
  mlp_forward<KT1>(hbuf, w1_s, w2_s, w3_s, b1, b2, b3, out);

  const int col = lane & 15, mo = (lane & 16) ? 8 : 0;
  float* yb = Y + ((size_t)tile * 16) * OUT_;
#pragma unroll
  for (int nt = 0; nt < 8; ++nt)
#pragma unroll
    for (int r = 0; r < 8; ++r)
      yb[(size_t)(r + mo) * OUT_ + (nt << 4) + col] = out[nt][r];
}

// ---------------------------------------------------------------------------
// Attention dots, LeakyReLU scores, per-batch softmax over 2048, combine,
// sigmoid. One workgroup per batch.
// ---------------------------------------------------------------------------
__global__ __launch_bounds__(256) void finalize_kernel(
    const float* __restrict__ attn, const float* __restrict__ pre_f1,
    const float* __restrict__ sub_f1, const float* __restrict__ self_f1,
    const float* __restrict__ ma_f2, float* __restrict__ out) {
  __shared__ float sc[4 * NOP_];
  __shared__ float attn_s[OUT_];
  __shared__ float red[256];
  const int b = blockIdx.x, tid = threadIdx.x;

  if (tid < OUT_) attn_s[tid] = attn[tid];
  __syncthreads();

  for (int op = tid; op < NOP_; op += 256) {
    size_t base = ((size_t)b * NOP_ + op) * OUT_;
    float d0 = 0.f, d1 = 0.f, d2 = 0.f, d3 = 0.f;
    for (int f = 0; f < OUT_; ++f) {
      float a = attn_s[f];
      d0 += pre_f1[base + f] * a;
      d1 += sub_f1[base + f] * a;
      d2 += self_f1[base + f] * a;
      d3 += ma_f2[base + f] * a;
    }
    auto lrelu = [](float x) { return x > 0.0f ? x : 0.2f * x; };
    sc[op]            = lrelu(d0 + d2);
    sc[NOP_ + op]     = lrelu(d1 + d2);
    sc[2 * NOP_ + op] = lrelu(d2 + d2);
    sc[3 * NOP_ + op] = lrelu(d3 + d2);
  }
  __syncthreads();

  float m = -3.4e38f;
  for (int i = tid; i < 4 * NOP_; i += 256) m = fmaxf(m, sc[i]);
  red[tid] = m;
  __syncthreads();
  for (int s = 128; s > 0; s >>= 1) {
    if (tid < s) red[tid] = fmaxf(red[tid], red[tid + s]);
    __syncthreads();
  }
  float mx = red[0];
  __syncthreads();

  float ssum = 0.f;
  for (int i = tid; i < 4 * NOP_; i += 256) {
    float e = expf(sc[i] - mx);
    sc[i] = e;
    ssum += e;
  }
  __syncthreads();
  red[tid] = ssum;
  __syncthreads();
  for (int s = 128; s > 0; s >>= 1) {
    if (tid < s) red[tid] += red[tid + s];
    __syncthreads();
  }
  float inv = 1.0f / red[0];
  __syncthreads();

  for (int idx = tid; idx < NOP_ * OUT_; idx += 256) {
    int op = idx >> 7, f = idx & (OUT_ - 1);
    size_t base = ((size_t)b * NOP_ + op) * OUT_ + f;
    float a0 = sc[op] * inv, a1 = sc[NOP_ + op] * inv;
    float a2 = sc[2 * NOP_ + op] * inv, a3 = sc[3 * NOP_ + op] * inv;
    float v = a0 * pre_f1[base] + a1 * sub_f1[base] +
              a2 * self_f1[base] + a3 * ma_f2[base];
    out[base] = 1.0f / (1.0f + expf(-v));
  }
}

// ---------------------------------------------------------------------------
extern "C" void kernel_launch(void* const* d_in, const int* in_sizes, int n_in,
                              void* d_out, int out_size, void* d_ws,
                              size_t ws_size, hipStream_t stream) {
  (void)in_sizes; (void)n_in; (void)out_size; (void)ws_size;
  const float* op      = (const float*)d_in[0];
  const float* ma      = (const float*)d_in[1];
  const float* eg      = (const float*)d_in[2];
  const int*   ma_adj  = (const int*)d_in[3];
  const int*   pre_adj = (const int*)d_in[4];   // batch_idxes == arange(B)
  const int*   sub_adj = (const int*)d_in[5];
  const float* attn    = (const float*)d_in[7];
  const float* W[5][6];
  for (int m = 0; m < 5; ++m)
    for (int t = 0; t < 6; ++t) W[m][t] = (const float*)d_in[8 + m * 6 + t];
  float* out = (float*)d_out;

  // --- workspace carve-up: f16 fragments first, then f32 intermediates ---
  _Float16* wsH = (_Float16*)d_ws;
  size_t off = 0, w1off[5], w2off[5], w3off[5];
  for (int m = 0; m < 5; ++m) {
    int kt1 = (m == 4) ? 4 : 1;
    w1off[m] = off; off += (size_t)kt1 * 16 * 512;
    w2off[m] = off; off += (size_t)8 * 16 * 512;
    w3off[m] = off; off += (size_t)8 * 8 * 512;
  }
  size_t opt_off = off; off += (size_t)B_ * 16 * 512;
  size_t halfBytes = (off * 2 + 255) & ~(size_t)255;
  float* wsF     = (float*)((char*)d_ws + halfBytes);
  float* pre_x   = wsF;                          // B*NOP*16
  float* sub_x   = pre_x + (size_t)B_ * NOP_ * 16;
  float* ma_in   = sub_x + (size_t)B_ * NOP_ * 16;   // B*NOP*128
  float* pre_f1  = ma_in  + (size_t)B_ * NOP_ * OUT_;
  float* sub_f1  = pre_f1 + (size_t)B_ * NOP_ * OUT_;
  float* self_f1 = sub_f1 + (size_t)B_ * NOP_ * OUT_;
  float* ma_f2   = self_f1 + (size_t)B_ * NOP_ * OUT_;

  // 1) fragment-swizzle all weights (f32 -> f16 B-operand tiles)
  for (int m = 0; m < 5; ++m) {
    int din = (m == 4) ? 128 : 16;
    int kp1 = (m == 4) ? 128 : 32;
    swizzle_weight<<<64, 256, 0, stream>>>(W[m][0], wsH + w1off[m], din, 256, kp1);
    swizzle_weight<<<64, 256, 0, stream>>>(W[m][2], wsH + w2off[m], 256, 256, 256);
    swizzle_weight<<<64, 256, 0, stream>>>(W[m][4], wsH + w3off[m], 256, 128, 256);
  }
  // per-batch op as B-operand tiles (K=512, N=16) for adjacency GEMMs
  for (int b = 0; b < B_; ++b)
    swizzle_weight<<<32, 256, 0, stream>>>(op + (size_t)b * NOP_ * 16,
                                           wsH + opt_off + (size_t)b * 16 * 512,
                                           512, 16, 512);

  // 2) adjacency GEMMs (WMMA over K=512, op-tiles DMA'd to LDS)
  gemm_adj_kernel<<<128, 128, 0, stream>>>(pre_adj, wsH + opt_off, pre_x);
  gemm_adj_kernel<<<128, 128, 0, stream>>>(sub_adj, wsH + opt_off, sub_x);

  // 3) dominant stage: mlp0 over 524288 rows + masked-mean reduction
  mlp0_agg_kernel<<<(B_ * NOP_) / 8, 256, 0, stream>>>(
      ma_adj, ma, eg, wsH + w1off[0], wsH + w2off[0], wsH + w3off[0],
      W[0][1], W[0][3], W[0][5], ma_in);

  // 4) mlp1..4
  mlp_rows_kernel<1, 8><<<64, 256, 0, stream>>>(pre_x, wsH + w1off[1],
      wsH + w2off[1], wsH + w3off[1], W[1][1], W[1][3], W[1][5], pre_f1, 16);
  mlp_rows_kernel<1, 8><<<64, 256, 0, stream>>>(sub_x, wsH + w1off[2],
      wsH + w2off[2], wsH + w3off[2], W[2][1], W[2][3], W[2][5], sub_f1, 16);
  mlp_rows_kernel<1, 8><<<64, 256, 0, stream>>>(op, wsH + w1off[3],
      wsH + w2off[3], wsH + w3off[3], W[3][1], W[3][3], W[3][5], self_f1, 16);
  mlp_rows_kernel<4, 4><<<128, 128, 0, stream>>>(ma_in, wsH + w1off[4],
      wsH + w2off[4], wsH + w3off[4], W[4][1], W[4][3], W[4][5], ma_f2, 128);

  // 5) scores / softmax / combine / sigmoid
  finalize_kernel<<<B_, 256, 0, stream>>>(attn, pre_f1, sub_f1, self_f1,
                                          ma_f2, out);
}
